// Generator_82197084110905
// MI455X (gfx1250) — matrix operand order repaired and checked by reading, not compiled, not run on hardware
//
#include <hip/hip_runtime.h>
#include <hip/hip_bf16.h>

// ---------------------------------------------------------------------------
// MI455X (gfx1250, wave32) implementation of the GAT sampler forward pass.
// Heavy GEMMs: v_wmma_f32_16x16x32_f16 with double-buffered
// global_load_async_to_lds_b128 staging (ASYNCcnt pipeline).
// Graph aggregation uses a destination-sorted CSR built once on device.
// ---------------------------------------------------------------------------

typedef __attribute__((ext_vector_type(16))) _Float16 v16h;
typedef __attribute__((ext_vector_type(8)))  _Float16 v8h;
typedef __attribute__((ext_vector_type(8)))  float    v8f;

union F16x16 { v16h v; v8h h[2]; };

#define IN_DIM   256
#define HID      256
#define NNODES   16
#define FEATD    64
#define NEDGES   5

// ------------------------------- GEMM --------------------------------------
// C[M,Nn] (f32) = A16[M,K] (f16 row-major) * Bt16[Nn,K] (f16, pre-transposed)
// Block: 256 threads = 8 waves; tile 128(M) x 64(N); K-step 32.
// Each wave owns a 32x32 output tile = 2x2 WMMA 16x16 tiles.
#define BM 128
#define BN 64
#define BK 32
#define SA 40   // LDS row stride (halves) for A tile, padded
#define SB 40   // LDS row stride (halves) for B tile, padded

// async copy: 32B (two b128) global -> LDS, ASYNCcnt-tracked.
// INST_OFFSET applies to both the LDS and global addresses (ISA 15.18.3).
__device__ __forceinline__ void async_g2l_32(unsigned lds_off, const void* gptr) {
  asm volatile("global_load_async_to_lds_b128 %0, %1, off\n\t"
               "global_load_async_to_lds_b128 %0, %1, off offset:16"
               :: "v"(lds_off), "v"((unsigned long long)(uintptr_t)gptr)
               : "memory");
}
__device__ __forceinline__ void async_g2l_16(unsigned lds_off, const void* gptr) {
  asm volatile("global_load_async_to_lds_b128 %0, %1, off"
               :: "v"(lds_off), "v"((unsigned long long)(uintptr_t)gptr)
               : "memory");
}

__global__ __launch_bounds__(256)
void gemm_f16_wmma(const _Float16* __restrict__ A, const _Float16* __restrict__ Bt,
                   float* __restrict__ C, int M, int Nn, int K) {
  __shared__ _Float16 sA[2][BM * SA];
  __shared__ _Float16 sB[2][BN * SB];

  const int tid   = threadIdx.x;
  const int lane  = tid & 31;
  const int wid   = tid >> 5;
  const int waveM = (wid >> 1) * 32;   // 0,32,64,96
  const int waveN = (wid & 1) * 32;    // 0,32
  const int blockM = blockIdx.x * BM;
  const int blockN = blockIdx.y * BN;

  v8f z = {0.f,0.f,0.f,0.f,0.f,0.f,0.f,0.f};
  v8f acc00 = z, acc01 = z, acc10 = z, acc11 = z;

  // staging assignments (uniform issue count: 3 async b128 per thread/tile)
  const int ar  = tid >> 1;                      // A row 0..127
  const int acb = (tid & 1) * 16;                // A col offset (halves)
  const int bn  = tid >> 2;                      // B row (n) 0..63
  const int bkf = (tid & 3) * 8;                 // B col offset (halves)
  const int growc = min(blockM + ar, M - 1);     // clamped (guarded at store)
  const int gnc   = min(blockN + bn, Nn - 1);

  const unsigned ldsA[2] = {
    (unsigned)(uintptr_t)&sA[0][ar * SA + acb],
    (unsigned)(uintptr_t)&sA[1][ar * SA + acb] };
  const unsigned ldsB[2] = {
    (unsigned)(uintptr_t)&sB[0][bn * SB + bkf],
    (unsigned)(uintptr_t)&sB[1][bn * SB + bkf] };

  const _Float16* gA = A  + (size_t)growc * K + acb;
  const _Float16* gB = Bt + (size_t)gnc   * K + bkf;

  const int l  = lane & 15;
  const int hi = lane >> 4;
  const int T  = K / BK;                 // K is a multiple of 32 for all calls

  // prologue: stage tile 0 into buffer 0
  async_g2l_32(ldsA[0], gA);
  async_g2l_16(ldsB[0], gB);

  for (int t = 0; t < T; ++t) {
    const int buf = t & 1;
    if (t + 1 < T) {   // issue tile t+1 into the other buffer, then wait for tile t
      const int k1 = (t + 1) * BK;
      async_g2l_32(ldsA[buf ^ 1], gA + k1);
      async_g2l_16(ldsB[buf ^ 1], gB + k1);
      asm volatile("s_wait_asynccnt 0x3" ::: "memory");
    } else {
      asm volatile("s_wait_asynccnt 0x0" ::: "memory");
    }
    __syncthreads();   // publish all waves' tile-t data

    // A fragments: lane<16 -> row l, K 0..7 & 16..23 ; lane>=16 -> K 8..15 & 24..31
    F16x16 af0, af1, bf0, bf1;
    {
      const _Float16* p = &sA[buf][(waveM + l) * SA + hi * 8];
      af0.h[0] = *(const v8h*)p;        af0.h[1] = *(const v8h*)(p + 16);
      p += 16 * SA;
      af1.h[0] = *(const v8h*)p;        af1.h[1] = *(const v8h*)(p + 16);
    }
    // B fragments: lane<16 -> col l, K 0..7 & 16..23 ; lane>=16 -> K 8..15 & 24..31
    {
      const _Float16* p = &sB[buf][(waveN + l) * SB + hi * 8];
      bf0.h[0] = *(const v8h*)p;        bf0.h[1] = *(const v8h*)(p + 16);
      p += 16 * SB;
      bf1.h[0] = *(const v8h*)p;        bf1.h[1] = *(const v8h*)(p + 16);
    }
    acc00 = __builtin_amdgcn_wmma_f32_16x16x32_f16(false, af0.v, false, bf0.v, (short)0, acc00, false, false);
    acc01 = __builtin_amdgcn_wmma_f32_16x16x32_f16(false, af0.v, false, bf1.v, (short)0, acc01, false, false);
    acc10 = __builtin_amdgcn_wmma_f32_16x16x32_f16(false, af1.v, false, bf0.v, (short)0, acc10, false, false);
    acc11 = __builtin_amdgcn_wmma_f32_16x16x32_f16(false, af1.v, false, bf1.v, (short)0, acc11, false, false);
    __syncthreads();   // buffer `buf` is refilled with tile t+2 at iteration t+1
  }

  // writeback: VGPR r holds rows (hi?8:0)+r, lane&15 is the column
  v8f accs[2][2] = {{acc00, acc01}, {acc10, acc11}};
  for (int mi = 0; mi < 2; ++mi)
    for (int ni = 0; ni < 2; ++ni) {
      const int col = blockN + waveN + ni * 16 + l;
      const int rb  = blockM + waveM + mi * 16 + hi * 8;
      if (col < Nn) {
        v8f c = accs[mi][ni];
#pragma unroll
        for (int r = 0; r < 8; ++r) {
          const int row = rb + r;
          if (row < M) C[(size_t)row * Nn + col] = c[r];
        }
      }
    }
}

// --------------------------- conversion kernels -----------------------------
__global__ void convert_f16(const float* __restrict__ in, _Float16* __restrict__ out, int n) {
  int i = blockIdx.x * blockDim.x + threadIdx.x;
  if (i < n) out[i] = (_Float16)in[i];
}

// out16[n*K + k] = (f16) W[k*N + n]   (weight transpose for WMMA B staging)
__global__ void convert_f16_T(const float* __restrict__ W, _Float16* __restrict__ out,
                              int K, int Nn) {
  int i = blockIdx.x * blockDim.x + threadIdx.x;
  if (i >= K * Nn) return;
  int n = i / K, k = i - n * K;
  out[i] = (_Float16)W[(size_t)k * Nn + n];
}

// ----------------------------- small dense ops ------------------------------
__global__ void tf_kernel(const float* __restrict__ feat, const float* __restrict__ Wt,
                          const float* __restrict__ bt, const int* __restrict__ tgt,
                          float* __restrict__ tf) {
  int j = blockIdx.x * blockDim.x + threadIdx.x;
  if (j >= 2 * HID) return;
  const float* f = feat + (size_t)tgt[0] * IN_DIM;
  float s = 0.f;
  for (int k = 0; k < IN_DIM; ++k) s += f[k] * Wt[(size_t)k * (2 * HID) + j];
  s += bt[j];
  tf[j] = s > 0.f ? s : 0.01f * s;
}

// el/er: out[i,h] = sum_d h[i,h,d] * a[h,d]
__global__ void attn_dot(const float* __restrict__ h, const float* __restrict__ av,
                         float* __restrict__ out, int N, int H, int D) {
  int i = blockIdx.x * blockDim.x + threadIdx.x;
  if (i >= N * H) return;
  int node = i / H, hd = i - node * H;
  const float* hp = h + (size_t)node * H * D + hd * D;
  const float* ap = av + hd * D;
  float s = 0.f;
  for (int d = 0; d < D; ++d) s += hp[d] * ap[d];
  out[i] = s;
}

// ------------------------------- CSR build ----------------------------------
__global__ void edge_count(const int* __restrict__ dst, int* __restrict__ cnt, int E) {
  int e = blockIdx.x * blockDim.x + threadIdx.x;
  if (e < E) atomicAdd(&cnt[dst[e]], 1);
}

__global__ __launch_bounds__(1024)
void scan_kernel(const int* __restrict__ cnt, int* __restrict__ off,
                 int* __restrict__ cur, int n) {
  __shared__ int sh[1024];
  __shared__ int carry;
  if (threadIdx.x == 0) carry = 0;
  __syncthreads();
  for (int base = 0; base < n; base += 1024) {
    int i = base + (int)threadIdx.x;
    int v = (i < n) ? cnt[i] : 0;
    sh[threadIdx.x] = v; __syncthreads();
    for (int o = 1; o < 1024; o <<= 1) {
      int t = (threadIdx.x >= (unsigned)o) ? sh[threadIdx.x - o] : 0;
      __syncthreads();
      sh[threadIdx.x] += t; __syncthreads();
    }
    int excl = sh[threadIdx.x] - v + carry;
    if (i < n) { off[i] = excl; cur[i] = excl; }
    __syncthreads();
    if (threadIdx.x == 1023) carry += sh[1023];
    __syncthreads();
  }
  if (threadIdx.x == 0) off[n] = carry;
}

__global__ void edge_scatter(const int* __restrict__ src, const int* __restrict__ dst,
                             int* __restrict__ cur, int* __restrict__ src_sorted, int E) {
  int e = blockIdx.x * blockDim.x + threadIdx.x;
  if (e >= E) return;
  int pos = atomicAdd(&cur[dst[e]], 1);
  src_sorted[pos] = src[e];
}

// --------------------------- GAT edge softmax + agg -------------------------
// One wave per (node, head). 3 passes over the node's edge list:
// max logit -> sum exp -> weighted gather of h[src]. out = agg + r + bias.
__global__ void gat_aggregate(const float* __restrict__ h, const float* __restrict__ r,
                              const float* __restrict__ bias,
                              const float* __restrict__ el, const float* __restrict__ er,
                              const int* __restrict__ off, const int* __restrict__ srcs,
                              float* __restrict__ out, int N, int H, int D) {
  int wave = (blockIdx.x * blockDim.x + threadIdx.x) >> 5;
  int lane = threadIdx.x & 31;
  if (wave >= N * H) return;
  int node = wave / H, hd = wave - node * H;
  int beg = off[node], end = off[node + 1];
  float eri = er[node * H + hd];

  float m = -1e30f;
  for (int p = beg; p < end; ++p) {
    float lg = el[srcs[p] * H + hd] + eri;
    lg = lg > 0.f ? lg : 0.2f * lg;
    m = fmaxf(m, lg);
  }
  float s = 0.f;
  for (int p = beg; p < end; ++p) {
    float lg = el[srcs[p] * H + hd] + eri;
    lg = lg > 0.f ? lg : 0.2f * lg;
    s += __expf(lg - m);
  }
  float inv = (s > 0.f) ? 1.f / s : 0.f;

  float acc[8];
#pragma unroll
  for (int q = 0; q < 8; ++q) acc[q] = 0.f;
  for (int p = beg; p < end; ++p) {
    int srow = srcs[p];
    float lg = el[srow * H + hd] + eri;
    lg = lg > 0.f ? lg : 0.2f * lg;
    float w = __expf(lg - m) * inv;
    const float* hp = h + (size_t)srow * H * D + hd * D;
    int q = 0;
    for (int c = lane; c < D; c += 32, ++q) acc[q] += w * hp[c];
  }
  float* op = out + (size_t)node * H * D + hd * D;
  const float* rp = r + (size_t)node * H * D + hd * D;
  const float* bp = bias + hd * D;
  int q = 0;
  for (int c = lane; c < D; c += 32, ++q) op[c] = acc[q] + rp[c] + bp[c];
}

// a2 = leaky(concat(a,e)+tf), e2 = leaky(concat(e,a)+tf) -> f16 for GEMM
__global__ void build_layer2(const float* __restrict__ a, const float* __restrict__ e,
                             const float* __restrict__ tf,
                             _Float16* __restrict__ a2, _Float16* __restrict__ e2, int N) {
  int idx = blockIdx.x * blockDim.x + threadIdx.x;
  if (idx >= N * 2 * HID) return;
  int i = idx >> 9, j = idx & 511;
  float t = tf[j];
  float va = (j < HID) ? a[(size_t)i * HID + j] : e[(size_t)i * HID + j - HID];
  float ve = (j < HID) ? e[(size_t)i * HID + j] : a[(size_t)i * HID + j - HID];
  float x = va + t; x = x > 0.f ? x : 0.01f * x;
  float y = ve + t; y = y > 0.f ? y : 0.01f * y;
  a2[idx] = (_Float16)x;
  e2[idx] = (_Float16)y;
}

// column partial sums of a3 [N,1024] -> attrsum[1024] (atomic over row chunks)
__global__ void colsum_kernel(const float* __restrict__ a3, float* __restrict__ attrsum,
                              int N, int rowsPer) {
  int col = blockIdx.x * blockDim.x + threadIdx.x;     // 0..1023
  int r0 = blockIdx.y * rowsPer;
  int r1 = min(r0 + rowsPer, N);
  float s = 0.f;
  for (int i = r0; i < r1; ++i) s += a3[(size_t)i * (NNODES * FEATD) + col];
  atomicAdd(&attrsum[col], s);
}

// node_attr = leaky(attr) @ Wo + bo ; written twice (outputs 0 and 1)
__global__ void node_attr_kernel(const float* __restrict__ attrsum, const float* __restrict__ Wo,
                                 const float* __restrict__ bo, float* __restrict__ out, float invN) {
  int j = threadIdx.x;                                  // 0..255
  for (int p = 0; p < NNODES; ++p) {
    float s = 0.f;
    for (int d = 0; d < FEATD; ++d) {
      float v = attrsum[p * FEATD + d] * invN;
      v = v > 0.f ? v : 0.01f * v;
      s += v * Wo[(size_t)d * IN_DIM + j];
    }
    s += bo[j];
    out[p * IN_DIM + j] = s;
    out[NNODES * IN_DIM + p * IN_DIM + j] = s;
  }
}

// probs[j, :] = softmax over nodes of e3[:, j]   (16 blocks)
__global__ void probs_kernel(const float* __restrict__ e3, float* __restrict__ probs, int N) {
  __shared__ float red[256];
  int j = blockIdx.x;
  float m = -1e30f;
  for (int i = threadIdx.x; i < N; i += 256) m = fmaxf(m, e3[(size_t)i * NNODES + j]);
  red[threadIdx.x] = m; __syncthreads();
  for (int o = 128; o; o >>= 1) {
    if (threadIdx.x < (unsigned)o) red[threadIdx.x] = fmaxf(red[threadIdx.x], red[threadIdx.x + o]);
    __syncthreads();
  }
  m = red[0]; __syncthreads();
  float s = 0.f;
  for (int i = threadIdx.x; i < N; i += 256) s += __expf(e3[(size_t)i * NNODES + j] - m);
  red[threadIdx.x] = s; __syncthreads();
  for (int o = 128; o; o >>= 1) {
    if (threadIdx.x < (unsigned)o) red[threadIdx.x] += red[threadIdx.x + o];
    __syncthreads();
  }
  s = red[0];
  float inv = 1.f / s;
  for (int i = threadIdx.x; i < N; i += 256)
    probs[(size_t)j * N + i] = __expf(e3[(size_t)i * NNODES + j] - m) * inv;
}

// Gumbel-max categorical sampling without replacement (deterministic hash RNG).
// One wave per output row; 16 waves in one block.
__device__ __forceinline__ unsigned mix_hash(unsigned x) {
  x ^= x >> 16; x *= 2246822519u; x ^= x >> 13; x *= 3266489917u; x ^= x >> 16;
  return x;
}
__global__ __launch_bounds__(512)
void sample_kernel(const float* __restrict__ probs, float* __restrict__ edges,
                   const int* __restrict__ tgt, int N) {
  int row  = threadIdx.x >> 5;
  int lane = threadIdx.x & 31;
  if (row >= NNODES) return;
  const float* p = probs + (size_t)row * N;
  int target = tgt[0];
  int chosen[NEDGES];
  chosen[0] = target;
  if (lane == 0) edges[row * NEDGES + 0] = (float)target;
  for (int k = 1; k < NEDGES; ++k) {
    float best = -1e30f; int bestIdx = 0x7fffffff;
    for (int i = lane; i < N; i += 32) {
      bool msk = false;
      for (int c = 0; c < k; ++c) msk |= (chosen[c] == i);
      if (msk) continue;
      unsigned hh = mix_hash((unsigned)i * 2654435761u ^ (unsigned)(row + 1) * 374761393u
                             ^ (unsigned)k * 668265263u);
      float u = fmaxf((hh >> 8) * (1.0f / 16777216.0f), 1e-7f);
      float g = -__logf(-__logf(u));
      float sc = p[i] + g;       // probs values as logits of masked softmax
      if (sc > best) { best = sc; bestIdx = i; }
    }
#pragma unroll
    for (int o = 16; o; o >>= 1) {
      float b2 = __shfl_xor(best, o, 32);
      int   i2 = __shfl_xor(bestIdx, o, 32);
      if (b2 > best || (b2 == best && i2 < bestIdx)) { best = b2; bestIdx = i2; }
    }
    chosen[k] = bestIdx;
    if (lane == 0) edges[row * NEDGES + k] = (float)bestIdx;
  }
}

// ------------------------------- launcher -----------------------------------
extern "C" void kernel_launch(void* const* d_in, const int* in_sizes, int n_in,
                              void* d_out, int out_size, void* d_ws, size_t ws_size,
                              hipStream_t stream) {
  const float* feat = (const float*)d_in[0];
  const int*   src  = (const int*)d_in[1];
  const int*   dst  = (const int*)d_in[2];
  const int*   tgt  = (const int*)d_in[3];
  const float* W1a = (const float*)d_in[4];  const float* al1a = (const float*)d_in[5];
  const float* ar1a = (const float*)d_in[6]; const float* R1a = (const float*)d_in[7];
  const float* b1a = (const float*)d_in[8];
  const float* W1e = (const float*)d_in[9];  const float* al1e = (const float*)d_in[10];
  const float* ar1e = (const float*)d_in[11];const float* R1e = (const float*)d_in[12];
  const float* b1e = (const float*)d_in[13];
  const float* W2a = (const float*)d_in[14]; const float* al2a = (const float*)d_in[15];
  const float* ar2a = (const float*)d_in[16];const float* R2a = (const float*)d_in[17];
  const float* b2a = (const float*)d_in[18];
  const float* W2e = (const float*)d_in[19]; const float* al2e = (const float*)d_in[20];
  const float* ar2e = (const float*)d_in[21];const float* R2e = (const float*)d_in[22];
  const float* b2e = (const float*)d_in[23];
  const float* Wt = (const float*)d_in[24];  const float* bt = (const float*)d_in[25];
  const float* Wo = (const float*)d_in[26];  const float* bo = (const float*)d_in[27];
  float* out = (float*)d_out;

  const int N  = in_sizes[0] / IN_DIM;    // 10000
  const int E  = in_sizes[1];             // 90000
  const int O2 = NNODES * FEATD;          // 1024

  // ---- scratch bump allocator ----
  char* ws = (char*)d_ws;
  size_t off = 0;
  auto alloc = [&](size_t bytes) -> void* {
    off = (off + 255) & ~(size_t)255;
    void* p = ws + off; off += bytes; return p;
  };
  _Float16* feat16 = (_Float16*)alloc((size_t)N * IN_DIM * 2);
  _Float16* w1a16 = (_Float16*)alloc((size_t)IN_DIM * HID * 2);
  _Float16* r1a16 = (_Float16*)alloc((size_t)IN_DIM * HID * 2);
  _Float16* w1e16 = (_Float16*)alloc((size_t)IN_DIM * HID * 2);
  _Float16* r1e16 = (_Float16*)alloc((size_t)IN_DIM * HID * 2);
  _Float16* w2a16 = (_Float16*)alloc((size_t)2 * HID * O2 * 2);
  _Float16* r2a16 = (_Float16*)alloc((size_t)2 * HID * O2 * 2);
  _Float16* w2e16 = (_Float16*)alloc((size_t)2 * HID * NNODES * 2);
  _Float16* r2e16 = (_Float16*)alloc((size_t)2 * HID * NNODES * 2);
  float* h1a = (float*)alloc((size_t)N * HID * 4);
  float* r1a = (float*)alloc((size_t)N * HID * 4);
  float* h1e = (float*)alloc((size_t)N * HID * 4);
  float* r1e = (float*)alloc((size_t)N * HID * 4);
  float* aout = (float*)alloc((size_t)N * HID * 4);
  float* eout = (float*)alloc((size_t)N * HID * 4);
  _Float16* a2h = (_Float16*)alloc((size_t)N * 2 * HID * 2);
  _Float16* e2h = (_Float16*)alloc((size_t)N * 2 * HID * 2);
  float* h2a = (float*)alloc((size_t)N * O2 * 4);
  float* r2a = (float*)alloc((size_t)N * O2 * 4);
  float* h2e = (float*)alloc((size_t)N * NNODES * 4);
  float* r2e = (float*)alloc((size_t)N * NNODES * 4);
  float* a3  = (float*)alloc((size_t)N * O2 * 4);
  float* e3  = (float*)alloc((size_t)N * NNODES * 4);
  float* el1a = (float*)alloc((size_t)N * 4 * 4); float* er1a = (float*)alloc((size_t)N * 4 * 4);
  float* el1e = (float*)alloc((size_t)N * 4 * 4); float* er1e = (float*)alloc((size_t)N * 4 * 4);
  float* el2a = (float*)alloc((size_t)N * 4 * 4); float* er2a = (float*)alloc((size_t)N * 4 * 4);
  float* el2e = (float*)alloc((size_t)N * 4);     float* er2e = (float*)alloc((size_t)N * 4);
  float* tf = (float*)alloc(2 * HID * 4);
  float* attrsum = (float*)alloc(O2 * 4);
  int* cnt = (int*)alloc((size_t)(N + 1) * 4);
  int* csr = (int*)alloc((size_t)(N + 1) * 4);
  int* cur = (int*)alloc((size_t)(N + 1) * 4);
  int* src_sorted = (int*)alloc((size_t)E * 4);
  (void)ws_size; (void)n_in; (void)out_size;

  // ---- conversions ----
  int nf = N * IN_DIM;
  convert_f16<<<(nf + 255) / 256, 256, 0, stream>>>(feat, feat16, nf);
  convert_f16_T<<<(IN_DIM * HID + 255) / 256, 256, 0, stream>>>(W1a, w1a16, IN_DIM, HID);
  convert_f16_T<<<(IN_DIM * HID + 255) / 256, 256, 0, stream>>>(R1a, r1a16, IN_DIM, HID);
  convert_f16_T<<<(IN_DIM * HID + 255) / 256, 256, 0, stream>>>(W1e, w1e16, IN_DIM, HID);
  convert_f16_T<<<(IN_DIM * HID + 255) / 256, 256, 0, stream>>>(R1e, r1e16, IN_DIM, HID);
  convert_f16_T<<<(2 * HID * O2 + 255) / 256, 256, 0, stream>>>(W2a, w2a16, 2 * HID, O2);
  convert_f16_T<<<(2 * HID * O2 + 255) / 256, 256, 0, stream>>>(R2a, r2a16, 2 * HID, O2);
  convert_f16_T<<<(2 * HID * NNODES + 255) / 256, 256, 0, stream>>>(W2e, w2e16, 2 * HID, NNODES);
  convert_f16_T<<<(2 * HID * NNODES + 255) / 256, 256, 0, stream>>>(R2e, r2e16, 2 * HID, NNODES);

  // ---- CSR build (independent of GEMMs) ----
  hipMemsetAsync(cnt, 0, (size_t)(N + 1) * 4, stream);
  edge_count<<<(E + 255) / 256, 256, 0, stream>>>(dst, cnt, E);
  scan_kernel<<<1, 1024, 0, stream>>>(cnt, csr, cur, N);
  edge_scatter<<<(E + 255) / 256, 256, 0, stream>>>(src, dst, cur, src_sorted, E);

  // ---- target feature vector ----
  tf_kernel<<<2, 256, 0, stream>>>(feat, Wt, bt, tgt, tf);

  // ---- layer-1 GEMMs (WMMA) ----
  dim3 g1((N + BM - 1) / BM, HID / BN);
  gemm_f16_wmma<<<g1, 256, 0, stream>>>(feat16, w1a16, h1a, N, HID, IN_DIM);
  gemm_f16_wmma<<<g1, 256, 0, stream>>>(feat16, r1a16, r1a, N, HID, IN_DIM);
  gemm_f16_wmma<<<g1, 256, 0, stream>>>(feat16, w1e16, h1e, N, HID, IN_DIM);
  gemm_f16_wmma<<<g1, 256, 0, stream>>>(feat16, r1e16, r1e, N, HID, IN_DIM);

  // ---- layer-1 attention + aggregation ----
  int nh1 = N * 4;
  attn_dot<<<(nh1 + 255) / 256, 256, 0, stream>>>(h1a, al1a, el1a, N, 4, HID / 4);
  attn_dot<<<(nh1 + 255) / 256, 256, 0, stream>>>(h1a, ar1a, er1a, N, 4, HID / 4);
  attn_dot<<<(nh1 + 255) / 256, 256, 0, stream>>>(h1e, al1e, el1e, N, 4, HID / 4);
  attn_dot<<<(nh1 + 255) / 256, 256, 0, stream>>>(h1e, ar1e, er1e, N, 4, HID / 4);
  int aggB1 = (nh1 * 32 + 255) / 256;
  gat_aggregate<<<aggB1, 256, 0, stream>>>(h1a, r1a, b1a, el1a, er1a, csr, src_sorted, aout, N, 4, HID / 4);
  gat_aggregate<<<aggB1, 256, 0, stream>>>(h1e, r1e, b1e, el1e, er1e, csr, src_sorted, eout, N, 4, HID / 4);

  // ---- layer-2 inputs ----
  build_layer2<<<(N * 2 * HID + 255) / 256, 256, 0, stream>>>(aout, eout, tf, a2h, e2h, N);

  // ---- layer-2 GEMMs (WMMA) ----
  dim3 g2a((N + BM - 1) / BM, O2 / BN);
  gemm_f16_wmma<<<g2a, 256, 0, stream>>>(a2h, w2a16, h2a, N, O2, 2 * HID);
  gemm_f16_wmma<<<g2a, 256, 0, stream>>>(a2h, r2a16, r2a, N, O2, 2 * HID);
  dim3 g2e((N + BM - 1) / BM, 1);
  gemm_f16_wmma<<<g2e, 256, 0, stream>>>(e2h, w2e16, h2e, N, NNODES, 2 * HID);
  gemm_f16_wmma<<<g2e, 256, 0, stream>>>(e2h, r2e16, r2e, N, NNODES, 2 * HID);

  // ---- layer-2 attention + aggregation ----
  attn_dot<<<(nh1 + 255) / 256, 256, 0, stream>>>(h2a, al2a, el2a, N, 4, O2 / 4);
  attn_dot<<<(nh1 + 255) / 256, 256, 0, stream>>>(h2a, ar2a, er2a, N, 4, O2 / 4);
  attn_dot<<<(N + 255) / 256, 256, 0, stream>>>(h2e, al2e, el2e, N, 1, NNODES);
  attn_dot<<<(N + 255) / 256, 256, 0, stream>>>(h2e, ar2e, er2e, N, 1, NNODES);
  gat_aggregate<<<aggB1, 256, 0, stream>>>(h2a, r2a, b2a, el2a, er2a, csr, src_sorted, a3, N, 4, O2 / 4);
  int aggB2 = (N * 32 + 255) / 256;
  gat_aggregate<<<aggB2, 256, 0, stream>>>(h2e, r2e, b2e, el2e, er2e, csr, src_sorted, e3, N, 1, NNODES);

  // ---- outputs ----
  hipMemsetAsync(attrsum, 0, O2 * 4, stream);
  int rowsPer = (N + 39) / 40;
  dim3 gc(O2 / 256, 40);
  colsum_kernel<<<gc, 256, 0, stream>>>(a3, attrsum, N, rowsPer);
  node_attr_kernel<<<1, 256, 0, stream>>>(attrsum, Wo, bo, out, 1.0f / (float)N);

  float* probs_out = out + 2 * NNODES * IN_DIM + NNODES * NEDGES;   // offset 8272
  float* edges_out = out + 2 * NNODES * IN_DIM;                     // offset 8192
  probs_kernel<<<NNODES, 256, 0, stream>>>(e3, probs_out, N);
  sample_kernel<<<1, 512, 0, stream>>>(probs_out, edges_out, tgt, N);
}
